// NTXentLoss_27547920236997
// MI455X (gfx1250) — matrix operand order repaired and checked by reading, not compiled, not run on hardware
//
#include <hip/hip_runtime.h>
#include <hip/hip_bf16.h>

typedef __attribute__((ext_vector_type(16))) __bf16 v16bf;
typedef __attribute__((ext_vector_type(8)))  __bf16 v8bf;
typedef __attribute__((ext_vector_type(4)))  __bf16 v4bf;
typedef __attribute__((ext_vector_type(8)))  float  v8f;

#define DIMD 128
#define NB   4096
#define NROW 8192          // 2B
#define NTILE 512          // NROW/16
// exp((2c)-2) = exp2(c*2log2e - 2log2e); logits bounded in [-2,2] -> fixed shift,
// argument in [-5.77, 0] -> raw v_exp_f32 is safe (no denorm fixup needed)
#define K1 ( 2.885390081777927f)   //  2*log2(e)
#define K2 (-2.885390081777927f)   // -2*log2(e)

static __device__ __forceinline__ v16bf combine8(v8bf lo, v8bf hi) {
  return __builtin_shufflevector(lo, hi, 0,1,2,3,4,5,6,7,8,9,10,11,12,13,14,15);
}

// ---------------- Kernel 1: L2-normalize rows, convert to bf16 reps ----------------
__global__ __launch_bounds__(256) void ntx_normalize(const float* __restrict__ p1,
                                                     const float* __restrict__ p2,
                                                     __bf16* __restrict__ reps) {
  int row  = (blockIdx.x * blockDim.x + threadIdx.x) >> 5;   // 0..8191
  int lane = threadIdx.x & 31;
  if (row >= NROW) return;
  const float* src = (row < NB) ? (p1 + (size_t)row * DIMD)
                                : (p2 + (size_t)(row - NB) * DIMD);
  float4 x = *(const float4*)(src + lane * 4);
  float ss = x.x*x.x + x.y*x.y + x.z*x.z + x.w*x.w;
#pragma unroll
  for (int mask = 16; mask >= 1; mask >>= 1)
    ss += __shfl_xor(ss, mask, 32);
  float nrm   = sqrtf(ss);
  float scale = 1.0f / fmaxf(nrm, 1e-12f);
  v4bf o;
  o[0] = (__bf16)(x.x * scale);
  o[1] = (__bf16)(x.y * scale);
  o[2] = (__bf16)(x.z * scale);
  o[3] = (__bf16)(x.w * scale);
  *(v4bf*)(reps + (size_t)row * DIMD + lane * 4) = o;
}

// ---------------- Kernel 2: WMMA GEMM sweep + fixed-shift sum-exp ----------------
__global__ __launch_bounds__(256) void ntx_wmma_lse(const __bf16* __restrict__ reps,
                                                    float* __restrict__ partial) {
  const int rowTile = blockIdx.x;          // 0..511
  const int rowBase = rowTile * 16;
  const int wave = threadIdx.x >> 5;       // 0..7
  const int lane = threadIdx.x & 31;
  const int n16  = lane & 15;
  const int hi   = lane >> 4;              // 0 or 1

  // ---- A tile (16 x 128) loaded once, directly in WMMA A layout ----
  v16bf A[4];
  {
    const size_t abase = (size_t)(rowBase + n16) * DIMD;
#pragma unroll
    for (int ks = 0; ks < 4; ++ks) {
      int kb = ks * 32;
      v8bf lo = *(const v8bf*)(reps + abase + kb + hi * 8);
      v8bf h8 = *(const v8bf*)(reps + abase + kb + 16 + hi * 8);
      A[ks] = combine8(lo, h8);
    }
  }

  bool onDiag[8];
#pragma unroll
  for (int r = 0; r < 8; ++r) onDiag[r] = ((r + hi * 8) == n16);

  float s[8], pos[8];
#pragma unroll
  for (int r = 0; r < 8; ++r) { s[r] = 0.0f; pos[r] = 0.0f; }

  const int partnerTile = (rowTile + 256) & (NTILE - 1);

  // B tile: lane n holds column colBase+n, K = ks*32 + hi*8 .. +8 and ks*32 + 16 + hi*8 .. +8
  auto loadB = [&](int ct, v8bf* b) {
    const size_t bbase = (size_t)(ct * 16 + n16) * DIMD + hi * 8;
#pragma unroll
    for (int ks = 0; ks < 4; ++ks) {
      b[2 * ks]     = *(const v8bf*)(reps + bbase + ks * 32);
      b[2 * ks + 1] = *(const v8bf*)(reps + bbase + ks * 32 + 16);
    }
  };

  auto process = [&](int ct, const v8bf* b) {
    v8f c = {};
#pragma unroll
    for (int ks = 0; ks < 4; ++ks)
      c = __builtin_amdgcn_wmma_f32_16x16x32_bf16(false, A[ks], false,
                                                  combine8(b[2*ks], b[2*ks+1]),
                                                  (short)0, c, false, false);
    float e[8];
#pragma unroll
    for (int r = 0; r < 8; ++r)
      e[r] = __builtin_amdgcn_exp2f(fmaf(c[r], K1, K2)); // raw v_exp_f32
    if (ct == rowTile) {                          // wave-uniform, taken once/sweep
#pragma unroll
      for (int r = 0; r < 8; ++r) e[r] = onDiag[r] ? 0.0f : e[r];   // mask sim[i,i]
    }
    if (ct == partnerTile) {                      // wave-uniform, taken once/sweep
#pragma unroll
      for (int r = 0; r < 8; ++r) pos[r] += onDiag[r] ? c[r] : 0.0f; // raw dot
    }
#pragma unroll
    for (int r = 0; r < 8; ++r) s[r] += e[r];
  };

  // ---- unroll-2 ping-pong pipeline over 64 column tiles per wave ----
  v8bf ba[8], bb[8];
  loadB(wave, ba);
  loadB(wave + 8, bb);
  for (int it = 0; it < 64; it += 2) {
    int ctA = wave + it * 8;
    int ctB = ctA + 8;
    process(ctA, ba);
    loadB((ctA + 16) & (NTILE - 1), ba);
    process(ctB, bb);
    loadB((ctB + 16) & (NTILE - 1), bb);
  }

  // ---- plain-sum reduce across the 16 lanes of each half-wave group ----
#pragma unroll
  for (int mask = 1; mask <= 8; mask <<= 1) {
#pragma unroll
    for (int r = 0; r < 8; ++r) {
      s[r]   += __shfl_xor(s[r],   mask, 32);
      pos[r] += __shfl_xor(pos[r], mask, 32);
    }
  }

  // ---- stage per-wave sums in LDS, merge across 8 waves ----
  __shared__ float ls[8][16], lp[8][16];
  if (n16 < 8) {
    int row = n16 + hi * 8;
    ls[wave][row] = s[n16];
    lp[wave][row] = pos[n16];
  }
  __syncthreads();

  if (threadIdx.x < 16) {
    int row = threadIdx.x;
    float S = 0.0f, P = 0.0f;
#pragma unroll
    for (int w = 0; w < 8; ++w) { S += ls[w][row]; P += lp[w][row]; }
    // lse = 2 + log(S); pos logit = 2*P
    float lossRow = (2.0f + __logf(S)) - 2.0f * P;
#pragma unroll
    for (int mask = 1; mask <= 8; mask <<= 1)
      lossRow += __shfl_xor(lossRow, mask, 32);
    if (threadIdx.x == 0) partial[rowTile] = lossRow;
  }
}

// ---------------- Kernel 3: deterministic reduction of 512 partials ----------------
__global__ __launch_bounds__(512) void ntx_reduce(const float* __restrict__ partial,
                                                  float* __restrict__ out) {
  __shared__ float buf[512];
  int tid = threadIdx.x;
  buf[tid] = partial[tid];
  __syncthreads();
#pragma unroll
  for (int off = 256; off > 0; off >>= 1) {
    if (tid < off) buf[tid] += buf[tid + off];
    __syncthreads();
  }
  if (tid == 0) out[0] = buf[0] / 16384.0f;        // sum over 2B rows / 16384 (4B)
}

extern "C" void kernel_launch(void* const* d_in, const int* in_sizes, int n_in,
                              void* d_out, int out_size, void* d_ws, size_t ws_size,
                              hipStream_t stream) {
  const float* p1 = (const float*)d_in[0];
  const float* p2 = (const float*)d_in[1];
  float* out = (float*)d_out;

  __bf16* reps   = (__bf16*)d_ws;                              // 8192*128*2 = 2 MiB
  float* partial = (float*)((char*)d_ws + (size_t)NROW * DIMD * sizeof(__bf16));

  ntx_normalize<<<(NROW * 32) / 256, 256, 0, stream>>>(p1, p2, reps);
  ntx_wmma_lse<<<NTILE, 256, 0, stream>>>(reps, partial);
  ntx_reduce<<<1, 512, 0, stream>>>(partial, out);
}
